// CLAMSelector_76493367542296
// MI455X (gfx1250) — matrix operand order repaired and checked by reading, not compiled, not run on hardware
//
#include <hip/hip_runtime.h>

// ---------------- problem constants ----------------
#define BDIM 4
#define NDIM 16384
#define DDIM 512
#define HDIM 256
#define CDIM 2
#define KSEL 11468   // int(16384*0.7) clamped to [128, 16384]
#define KINST 8

typedef __attribute__((ext_vector_type(16))) __bf16 v16bf;
typedef __attribute__((ext_vector_type(8)))  float  v8f;
typedef unsigned int u32x4 __attribute__((ext_vector_type(4)));
typedef int          i32x4 __attribute__((ext_vector_type(4)));
typedef int          i32x8 __attribute__((ext_vector_type(8)));

union Frag { v16bf v; uint4 q[2]; };

__device__ __forceinline__ unsigned short f2bf(float f) {
  unsigned int u = __float_as_uint(f);
  u += 0x7FFFu + ((u >> 16) & 1u);           // round-to-nearest-even
  return (unsigned short)(u >> 16);
}
__device__ __forceinline__ unsigned int pk2(float a, float b) {
  return (unsigned int)f2bf(a) | ((unsigned int)f2bf(b) << 16);
}
__device__ __forceinline__ unsigned int fkey(float x) {   // order-preserving uint key
  unsigned int u = __float_as_uint(x);
  return (u & 0x80000000u) ? ~u : (u | 0x80000000u);
}

// ---------------- kernel 0: weight preconversion ----------------
// WaT[h][d] = bf16(Wa[d][h]);  W1T[c][h][d] = bf16(W1[c][d][h])
__global__ __launch_bounds__(256) void prep(const float* __restrict__ Wa,
                                            const float* __restrict__ W1,
                                            unsigned short* __restrict__ WaT,
                                            unsigned short* __restrict__ W1T) {
  int i = blockIdx.x * 256 + threadIdx.x;     // 0 .. 393215
  if (i < HDIM * DDIM) {
    int h = i >> 9, d = i & 511;
    WaT[i] = f2bf(Wa[d * HDIM + h]);
  } else {
    int j = i - HDIM * DDIM;                  // 0 .. 262143
    int c = j >> 17, r = j & 131071;
    int h = r >> 9, d = r & 511;
    W1T[j] = f2bf(W1[((size_t)c * DDIM + d) * HDIM + h]);
  }
}

// ---------------- kernel 1: fused attention GEMM ----------------
// a_logits[b][c][n] = relu(features[b,n,:] @ Wa + ba) . Wbr[c] + bbr[c]
// B tile (bf16 WaT K-slab) is staged into LDS by the Tensor Data Mover with
// hardware row padding (64B rows padded to 80B -> Bs[256][40]).
__global__ __launch_bounds__(256) void attn_gemm(const float* __restrict__ feat,
                                                 const unsigned short* __restrict__ WaT,
                                                 const float* __restrict__ ba,
                                                 const float* __restrict__ Wbr,
                                                 const float* __restrict__ bbr,
                                                 float* __restrict__ alog) {
  __shared__ __align__(16) unsigned short As[128][40];   // 128 rows x 32 K (bf16), +8 pad
  __shared__ __align__(16) unsigned short Bs[256][40];   // 256 h   x 32 K (bf16), +8 pad (TDM pad)
  const int t    = threadIdx.x;
  const int lane = t & 31, wv = t >> 5;
  const int half = lane >> 4, nlo = lane & 15;
  const int row0 = blockIdx.x * 128;          // global row over B*N
  const int b    = row0 >> 14;
  const int n0   = row0 & 16383;

  v8f zero;
  #pragma unroll
  for (int i = 0; i < 8; ++i) zero[i] = 0.0f;
  v8f acc[16];
  #pragma unroll
  for (int nt = 0; nt < 16; ++nt) acc[nt] = zero;

  const int arow = t >> 1, aseg = t & 1;
  for (int kk = 0; kk < 16; ++kk) {
    __syncthreads();
    { // A tile: 128x32 fp32 -> bf16 (VALU convert, LDS store)
      const float4* ap = (const float4*)(feat + (size_t)(row0 + arow) * DDIM + kk * 32 + aseg * 16);
      if (kk < 15) __builtin_prefetch((const void*)(ap + 8), 0, 0);  // next K slab
      float4 x0 = ap[0], x1 = ap[1], x2 = ap[2], x3 = ap[3];
      uint4 q0 = { pk2(x0.x, x0.y), pk2(x0.z, x0.w), pk2(x1.x, x1.y), pk2(x1.z, x1.w) };
      uint4 q1 = { pk2(x2.x, x2.y), pk2(x2.z, x2.w), pk2(x3.x, x3.y), pk2(x3.z, x3.w) };
      *(uint4*)&As[arow][aseg * 16]     = q0;
      *(uint4*)&As[arow][aseg * 16 + 8] = q1;
    }
    if (wv == 0) {  // wave-uniform: wave 0 drives the TDM for the B tile
      // ---- Tensor DMA descriptor (D#), 2D tile: 256 rows x 32 bf16 ----
      unsigned long long ga = (unsigned long long)(const void*)WaT + (unsigned long long)kk * 64ull;
      u32x4 g0;
      g0[0] = 1u;                                   // count=1, user mode, no gather
      g0[1] = (unsigned int)(size_t)&Bs[0][0];      // lds_addr (byte offset in LDS)
      g0[2] = (unsigned int)(ga & 0xFFFFFFFFu);     // global_addr[31:0]
      g0[3] = (unsigned int)(ga >> 32) | (2u << 30);// global_addr[56:32] | type=2 (image)
      i32x8 g1;
      g1[0] = (int)((1u << 16)      // data_size = 1 -> 2 bytes
                  | (1u << 20)      // pad_enable
                  | (3u << 22)      // pad_interval: every 16 DWORDs (one 64B row)
                  | (3u << 25));    // pad_amount: 4 DWORDs (16B) -> 80B row stride
      g1[1] = (int)(32u << 16);     // tensor_dim0 = 32  (bits 79:48, low half here)
      g1[2] = (int)(256u << 16);    // tensor_dim1 = 256 (bits 111:80, low half here)
      g1[3] = (int)(32u << 16);     // tile_dim0 = 32    (bits 127:112)
      g1[4] = 256;                  // tile_dim1 = 256   (bits 143:128); tile_dim2 = 0
      g1[5] = 512;                  // tensor_dim0_stride = 512 elements (bits 207:160)
      g1[6] = 0;
      g1[7] = 0;
      i32x4 gz; gz[0] = 0; gz[1] = 0; gz[2] = 0; gz[3] = 0;   // groups 2/3 unused (2D)
      i32x8 gz8;
      #pragma unroll
      for (int z = 0; z < 8; ++z) gz8[z] = 0;
      __builtin_amdgcn_tensor_load_to_lds(g0, g1, gz, gz, gz8, 0);
      __builtin_amdgcn_s_wait_tensorcnt(0);
    }
    __syncthreads();

    Frag a;
    const int m = wv * 16 + nlo;
    a.q[0] = *(const uint4*)&As[m][half * 8];        // K = half*8 + 0..7
    a.q[1] = *(const uint4*)&As[m][16 + half * 8];   // K = 16 + half*8 + 0..7
    #pragma unroll
    for (int g = 0; g < 4; ++g) {                    // preload 4 B frags, then 4 WMMAs
      Frag bf[4];
      #pragma unroll
      for (int j = 0; j < 4; ++j) {
        const int nt = g * 4 + j;
        bf[j].q[0] = *(const uint4*)&Bs[nt * 16 + nlo][half * 8];
        bf[j].q[1] = *(const uint4*)&Bs[nt * 16 + nlo][16 + half * 8];
      }
      #pragma unroll
      for (int j = 0; j < 4; ++j) {
        acc[g * 4 + j] = __builtin_amdgcn_wmma_f32_16x16x32_bf16(false, a.v, false, bf[j].v,
                                                                 (short)0, acc[g * 4 + j],
                                                                 false, false);
      }
    }
  }

  // epilogue: bias + relu, fused dot with Wbr rows, cross-lane reduce
  float l0[8], l1[8];
  #pragma unroll
  for (int v = 0; v < 8; ++v) { l0[v] = 0.0f; l1[v] = 0.0f; }
  #pragma unroll
  for (int nt = 0; nt < 16; ++nt) {
    const int col = nt * 16 + nlo;
    const float bav = ba[col];
    const float w0  = Wbr[col];
    const float w1  = Wbr[HDIM + col];
    #pragma unroll
    for (int v = 0; v < 8; ++v) {
      float hv = fmaxf(acc[nt][v] + bav, 0.0f);
      l0[v] += hv * w0;
      l1[v] += hv * w1;
    }
  }
  #pragma unroll
  for (int v = 0; v < 8; ++v) {
    #pragma unroll
    for (int s = 1; s < 16; s <<= 1) {
      l0[v] += __shfl_xor(l0[v], s, 32);
      l1[v] += __shfl_xor(l1[v], s, 32);
    }
  }
  if (nlo == 0) {
    const float bb0 = bbr[0], bb1 = bbr[1];
    #pragma unroll
    for (int v = 0; v < 8; ++v) {
      int n = n0 + wv * 16 + v + half * 8;   // C/D layout: M = v (+8 for upper half)
      alog[((size_t)b * CDIM + 0) * NDIM + n] = l0[v] + bb0;
      alog[((size_t)b * CDIM + 1) * NDIM + n] = l1[v] + bb1;
    }
  }
}

// ---------------- kernel 2: softmax stats (max, sumexp) per (b,c) ----------------
__global__ __launch_bounds__(256) void softmax_stats(const float* __restrict__ alog,
                                                     float* __restrict__ maxsum,
                                                     float* __restrict__ d_loss) {
  const int bc = blockIdx.x, tid = threadIdx.x;
  if (bc == 0 && tid == 0) *d_loss = 0.0f;    // zero loss before inst_loss kernel
  const float* p = alog + (size_t)bc * NDIM;
  __shared__ float red[256];
  float mx = -3.4e38f;
  for (int i = tid; i < NDIM; i += 256) mx = fmaxf(mx, p[i]);
  red[tid] = mx; __syncthreads();
  for (int s = 128; s > 0; s >>= 1) { if (tid < s) red[tid] = fmaxf(red[tid], red[tid + s]); __syncthreads(); }
  mx = red[0]; __syncthreads();
  float sm = 0.0f;
  for (int i = tid; i < NDIM; i += 256) sm += expf(p[i] - mx);
  red[tid] = sm; __syncthreads();
  for (int s = 128; s > 0; s >>= 1) { if (tid < s) red[tid] += red[tid + s]; __syncthreads(); }
  if (tid == 0) { maxsum[bc * 2] = mx; maxsum[bc * 2 + 1] = red[0]; }
}

// ---------------- kernel 3: combined = mean_c softmax ----------------
__global__ __launch_bounds__(256) void combined_k(const float* __restrict__ alog,
                                                  const float* __restrict__ maxsum,
                                                  float* __restrict__ comb) {
  int i = blockIdx.x * 256 + threadIdx.x;      // over B*N
  int b = i >> 14, n = i & 16383;
  float m0 = maxsum[(b * 2) * 2],     s0 = maxsum[(b * 2) * 2 + 1];
  float m1 = maxsum[(b * 2 + 1) * 2], s1 = maxsum[(b * 2 + 1) * 2 + 1];
  float a0 = expf(alog[((size_t)b * 2)     * NDIM + n] - m0) / s0;
  float a1 = expf(alog[((size_t)b * 2 + 1) * NDIM + n] - m1) / s1;
  comb[i] = 0.5f * (a0 + a1);
}

// ---------------- kernel 4: exact k-th largest via 4-pass byte radix ----------------
__global__ __launch_bounds__(256) void find_tau(const float* __restrict__ comb,
                                                uint2* __restrict__ tauneed) {
  const int b = blockIdx.x, tid = threadIdx.x;
  __shared__ unsigned int hist[256];
  __shared__ unsigned int s_prefix, s_remaining;
  if (tid == 0) { s_prefix = 0u; s_remaining = KSEL; }
  for (int pass = 3; pass >= 0; --pass) {
    __syncthreads();
    hist[tid] = 0; __syncthreads();
    const int shift = pass * 8;
    const unsigned int maskHi = (pass == 3) ? 0u : (0xFFFFFFFFu << ((pass + 1) * 8));
    const unsigned int pref = s_prefix;
    for (int n = tid; n < NDIM; n += 256) {
      unsigned int u = fkey(comb[(size_t)b * NDIM + n]);
      if ((u & maskHi) == (pref & maskHi))
        atomicAdd(&hist[(u >> shift) & 255u], 1u);
    }
    __syncthreads();
    if (tid == 0) {
      unsigned int rem = s_remaining, acc = 0;
      int bin = 255;
      for (; bin >= 0; --bin) { if (acc + hist[bin] >= rem) break; acc += hist[bin]; }
      if (bin < 0) bin = 0;
      s_prefix |= ((unsigned int)bin) << shift;
      s_remaining = rem - acc;
    }
  }
  __syncthreads();
  if (tid == 0) tauneed[b] = make_uint2(s_prefix, s_remaining);
}

// ---------------- kernel 5: instance classifier loss (WMMA, one wave) ----------------
__global__ __launch_bounds__(256) void inst_loss(const float* __restrict__ alog,
                                                 const float* __restrict__ feat,
                                                 const unsigned short* __restrict__ W1T,
                                                 const float* __restrict__ b1,
                                                 const float* __restrict__ W2,
                                                 const float* __restrict__ b2,
                                                 float* __restrict__ d_loss) {
  const int bc = blockIdx.x, b = bc >> 1, c = bc & 1;
  const int tid = threadIdx.x;
  __shared__ float cv[2048];
  __shared__ int   ci[2048];
  __shared__ int   fidx[16];
  __shared__ __align__(16) unsigned short instA[16][520];  // 16 rows x 512 bf16 (+8 pad)
  const float* lg = alog + (size_t)bc * NDIM;

  // ---- top-8 (attn order == logits order) ----
  {
    float bv[8]; int bi_[8];
    #pragma unroll
    for (int s = 0; s < 8; ++s) { bv[s] = -3.4e38f; bi_[s] = 0x7FFFFFFF; }
    for (int n = tid; n < NDIM; n += 256) {
      float v = lg[n]; int idx = n;
      #pragma unroll
      for (int s = 0; s < 8; ++s) {
        bool better = (v > bv[s]) || (v == bv[s] && idx < bi_[s]);
        float tv = bv[s]; int ti = bi_[s];
        if (better) { bv[s] = v; bi_[s] = idx; v = tv; idx = ti; }
      }
    }
    #pragma unroll
    for (int s = 0; s < 8; ++s) { cv[tid * 8 + s] = bv[s]; ci[tid * 8 + s] = bi_[s]; }
  }
  __syncthreads();
  if (tid == 0) {
    float bv[8]; int bi_[8];
    for (int s = 0; s < 8; ++s) { bv[s] = -3.4e38f; bi_[s] = 0x7FFFFFFF; }
    for (int i = 0; i < 2048; ++i) {
      float v = cv[i]; int idx = ci[i];
      for (int s = 0; s < 8; ++s) {
        bool better = (v > bv[s]) || (v == bv[s] && idx < bi_[s]);
        float tv = bv[s]; int ti = bi_[s];
        if (better) { bv[s] = v; bi_[s] = idx; v = tv; idx = ti; }
      }
    }
    for (int s = 0; s < 8; ++s) fidx[s] = bi_[s];
  }
  __syncthreads();
  // ---- bottom-8 ----
  {
    float bv[8]; int bi_[8];
    #pragma unroll
    for (int s = 0; s < 8; ++s) { bv[s] = 3.4e38f; bi_[s] = 0x7FFFFFFF; }
    for (int n = tid; n < NDIM; n += 256) {
      float v = lg[n]; int idx = n;
      #pragma unroll
      for (int s = 0; s < 8; ++s) {
        bool better = (v < bv[s]) || (v == bv[s] && idx < bi_[s]);
        float tv = bv[s]; int ti = bi_[s];
        if (better) { bv[s] = v; bi_[s] = idx; v = tv; idx = ti; }
      }
    }
    #pragma unroll
    for (int s = 0; s < 8; ++s) { cv[tid * 8 + s] = bv[s]; ci[tid * 8 + s] = bi_[s]; }
  }
  __syncthreads();
  if (tid == 0) {
    float bv[8]; int bi_[8];
    for (int s = 0; s < 8; ++s) { bv[s] = 3.4e38f; bi_[s] = 0x7FFFFFFF; }
    for (int i = 0; i < 2048; ++i) {
      float v = cv[i]; int idx = ci[i];
      for (int s = 0; s < 8; ++s) {
        bool better = (v < bv[s]) || (v == bv[s] && idx < bi_[s]);
        float tv = bv[s]; int ti = bi_[s];
        if (better) { bv[s] = v; bi_[s] = idx; v = tv; idx = ti; }
      }
    }
    for (int s = 0; s < 8; ++s) fidx[8 + s] = bi_[s];
  }
  __syncthreads();

  // ---- gather 16 instance rows into LDS as bf16 ----
  {
    const int row = tid >> 4, c0 = (tid & 15) * 32;
    const int n = fidx[row];
    const float4* src = (const float4*)(feat + ((size_t)b * NDIM + n) * DDIM + c0);
    #pragma unroll
    for (int g = 0; g < 4; ++g) {
      float4 x0 = src[g * 2 + 0];
      float4 x1 = src[g * 2 + 1];
      uint4 q = { pk2(x0.x, x0.y), pk2(x0.z, x0.w), pk2(x1.x, x1.y), pk2(x1.z, x1.w) };
      *(uint4*)&instA[row][c0 + g * 8] = q;
    }
  }
  __syncthreads();

  // ---- wave 0: 16x256 = inst(16x512) @ W1[c] via WMMA, then W2 + CE ----
  if (tid < 32) {
    const int lane = tid, half = lane >> 4, nlo = lane & 15;
    v8f zero;
    #pragma unroll
    for (int i = 0; i < 8; ++i) zero[i] = 0.0f;
    v8f acc[16];
    #pragma unroll
    for (int nt = 0; nt < 16; ++nt) acc[nt] = zero;
    for (int k = 0; k < 16; ++k) {
      Frag a;
      a.q[0] = *(const uint4*)&instA[nlo][k * 32 + half * 8];
      a.q[1] = *(const uint4*)&instA[nlo][k * 32 + 16 + half * 8];
      #pragma unroll
      for (int g = 0; g < 4; ++g) {
        Frag bf[4];
        #pragma unroll
        for (int j = 0; j < 4; ++j) {
          const int nt = g * 4 + j;
          const unsigned short* wp = W1T + ((size_t)c * HDIM + nt * 16 + nlo) * DDIM + k * 32 + half * 8;
          bf[j].q[0] = *(const uint4*)wp;
          bf[j].q[1] = *(const uint4*)(wp + 16);
        }
        #pragma unroll
        for (int j = 0; j < 4; ++j) {
          acc[g * 4 + j] = __builtin_amdgcn_wmma_f32_16x16x32_bf16(false, a.v, false, bf[j].v,
                                                                   (short)0, acc[g * 4 + j],
                                                                   false, false);
        }
      }
    }
    float o0[8], o1[8];
    #pragma unroll
    for (int v = 0; v < 8; ++v) { o0[v] = 0.0f; o1[v] = 0.0f; }
    #pragma unroll
    for (int nt = 0; nt < 16; ++nt) {
      const int col = nt * 16 + nlo;
      const float b1v = b1[c * HDIM + col];
      const float w20 = W2[((size_t)c * HDIM + col) * 2 + 0];
      const float w21 = W2[((size_t)c * HDIM + col) * 2 + 1];
      #pragma unroll
      for (int v = 0; v < 8; ++v) {
        float hv = fmaxf(acc[nt][v] + b1v, 0.0f);
        o0[v] += hv * w20;
        o1[v] += hv * w21;
      }
    }
    #pragma unroll
    for (int v = 0; v < 8; ++v) {
      #pragma unroll
      for (int s = 1; s < 16; s <<= 1) {
        o0[v] += __shfl_xor(o0[v], s, 32);
        o1[v] += __shfl_xor(o1[v], s, 32);
      }
    }
    float part = 0.0f;
    if (nlo == 0) {
      const float bb0 = b2[c * 2 + 0], bb1 = b2[c * 2 + 1];
      #pragma unroll
      for (int v = 0; v < 8; ++v) {
        int r = v + half * 8;                        // instance row 0..15
        float L0 = o0[v] + bb0, L1 = o1[v] + bb1;
        float mm = fmaxf(L0, L1);
        float lse = mm + logf(expf(L0 - mm) + expf(L1 - mm));
        part += lse - ((r < KINST) ? L1 : L0);       // label 1 for top rows
      }
    }
    part += __shfl_xor(part, 16, 32);
    if (lane == 0) atomicAdd(d_loss, part * (1.0f / (16.0f * (float)(BDIM * CDIM))));
  }
}

// ---------------- kernel 6: build sorted selection indices ----------------
__device__ __forceinline__ unsigned int block_scan_inc(unsigned int v,
                                                       volatile unsigned int* buf, int tid) {
  buf[tid] = v; __syncthreads();
  for (int off = 1; off < 256; off <<= 1) {
    unsigned int t = 0;
    if (tid >= off) t = buf[tid - off];
    __syncthreads();
    buf[tid] += t;
    __syncthreads();
  }
  return buf[tid];
}

__global__ __launch_bounds__(256) void build_sel(const float* __restrict__ comb,
                                                 const uint2* __restrict__ tn,
                                                 int* __restrict__ sidx,
                                                 float* __restrict__ sidx_f) {
  const int b = blockIdx.x, tid = threadIdx.x;
  const unsigned int tau = tn[b].x, need = tn[b].y;
  __shared__ unsigned int buf[256];
  __shared__ unsigned int run_sel, run_eq;
  if (tid == 0) { run_sel = 0; run_eq = 0; }
  __syncthreads();
  for (int base = 0; base < NDIM; base += 256) {
    const int n = base + tid;
    const unsigned int u = fkey(comb[(size_t)b * NDIM + n]);
    const unsigned int isgt = (u > tau) ? 1u : 0u;
    const unsigned int iseq = (u == tau) ? 1u : 0u;
    const unsigned int rs = run_sel, re = run_eq;
    unsigned int eqinc = block_scan_inc(iseq, buf, tid);
    unsigned int eqrank = re + eqinc - iseq;
    unsigned int take = isgt | (iseq & ((eqrank < need) ? 1u : 0u));
    __syncthreads();
    unsigned int tinc = block_scan_inc(take, buf, tid);
    if (take) {
      unsigned int pos = rs + tinc - take;
      if (pos < KSEL) {
        sidx[(size_t)b * KSEL + pos]   = n;
        sidx_f[(size_t)b * KSEL + pos] = (float)n;
      }
    }
    __syncthreads();
    if (tid == 255) { run_sel = rs + tinc; run_eq = re + eqinc; }
    __syncthreads();
  }
}

// ---------------- kernel 7: gather selected feature rows ----------------
__global__ __launch_bounds__(128) void gather_sel(const float* __restrict__ feat,
                                                  const int* __restrict__ sidx,
                                                  float* __restrict__ out_sel) {
  const int bid = blockIdx.x;
  const int b = bid / KSEL, j = bid % KSEL;
  const int n = sidx[(size_t)b * KSEL + j];
  const float4* src = (const float4*)(feat + ((size_t)b * NDIM + n) * DDIM);
  float4* dst = (float4*)(out_sel + ((size_t)b * KSEL + j) * DDIM);
  dst[threadIdx.x] = src[threadIdx.x];   // 128 lanes x 16B = 2 KiB row
}

// ---------------- host launcher ----------------
extern "C" void kernel_launch(void* const* d_in, const int* in_sizes, int n_in,
                              void* d_out, int out_size, void* d_ws, size_t ws_size,
                              hipStream_t stream) {
  (void)in_sizes; (void)n_in; (void)out_size; (void)ws_size;
  const float* feat = (const float*)d_in[0];
  const float* Wa   = (const float*)d_in[1];
  const float* ba   = (const float*)d_in[2];
  const float* Wbr  = (const float*)d_in[3];
  const float* bbr  = (const float*)d_in[4];
  const float* W1   = (const float*)d_in[5];
  const float* b1   = (const float*)d_in[6];
  const float* W2   = (const float*)d_in[7];
  const float* b2   = (const float*)d_in[8];

  // workspace layout (~1.43 MiB total)
  char* w = (char*)d_ws;
  unsigned short* WaT = (unsigned short*)(w);             // 256 KiB
  unsigned short* W1T = (unsigned short*)(w + 262144);    // 512 KiB
  float* alog         = (float*)(w + 786432);             // 512 KiB  [B][C][N]
  float* maxsum       = (float*)(w + 1310720);            // 64 B
  uint2* tauneed      = (uint2*)(w + 1310976);            // 32 B
  int*   sidx         = (int*)  (w + 1311232);            // 179.2 KiB

  // output layout: [selected | combined | sel_idx | cl_loss], all as float
  float* out      = (float*)d_out;
  float* out_sel  = out;
  float* out_comb = out + (size_t)BDIM * KSEL * DDIM;
  float* out_idx  = out_comb + (size_t)BDIM * NDIM;
  float* out_loss = out_idx + (size_t)BDIM * KSEL;

  prep<<<1536, 256, 0, stream>>>(Wa, W1, WaT, W1T);
  attn_gemm<<<(BDIM * NDIM) / 128, 256, 0, stream>>>(feat, WaT, ba, Wbr, bbr, alog);
  softmax_stats<<<BDIM * CDIM, 256, 0, stream>>>(alog, maxsum, out_loss);
  combined_k<<<(BDIM * NDIM) / 256, 256, 0, stream>>>(alog, maxsum, out_comb);
  find_tau<<<BDIM, 256, 0, stream>>>(out_comb, tauneed);
  inst_loss<<<BDIM * CDIM, 256, 0, stream>>>(alog, feat, W1T, b1, W2, b2, out_loss);
  build_sel<<<BDIM, 256, 0, stream>>>(out_comb, tauneed, sidx, out_idx);
  gather_sel<<<BDIM * KSEL, 128, 0, stream>>>(feat, sidx, out_sel);
}
